// Encoder_1486058684645
// MI455X (gfx1250) — matrix operand order repaired and checked by reading, not compiled, not run on hardware
//
#include <hip/hip_runtime.h>
#include <hip/hip_fp16.h>
#include <math.h>

typedef __attribute__((ext_vector_type(16))) _Float16 v16h;
typedef __attribute__((ext_vector_type(8)))  float    v8f;
typedef _Float16 f16;

#define HH 256
#define WWID 256
#define HW 65536
#define BB 16
#define CCH 64
#define LL 256
#define EPSV 1e-5f
#define USE_ASYNC 1

// async global->LDS copy (gfx1250, tracked by ASYNCcnt)
__device__ __forceinline__ void async_g2l_b32(unsigned lds_off, const float* gptr) {
#if USE_ASYNC
  asm volatile("global_load_async_to_lds_b32 %0, %1, off"
               :: "v"(lds_off), "v"(gptr) : "memory");
#endif
}
__device__ __forceinline__ void wait_async0() {
#if USE_ASYNC
  asm volatile("s_wait_asynccnt 0x0" ::: "memory");
#endif
}

// ---------------------------------------------------------------------------
// Watermark index generation (replicates _watermark_indices; all in channel 0)
// ---------------------------------------------------------------------------
__global__ void wm_idx(int* __restrict__ ys, int* __restrict__ xs) {
  if (threadIdx.x == 0 && blockIdx.x == 0) {
    int n = 0;
    for (int i = -10; i <= 10 && n < 256; ++i)
      for (int j = -10; j <= 10 && n < 256; ++j)
        if (i * i + j * j <= 100) { ys[n] = 128 + i; xs[n] = 128 + j; ++n; }
  }
}

// ---------------------------------------------------------------------------
// Weight packing: f32 -> f16, K-padded GEMM layout [co][Kpad]
// ---------------------------------------------------------------------------
__global__ void pack_w0(const float* __restrict__ w0, f16* __restrict__ w016) {
  int e = blockIdx.x * 256 + threadIdx.x;           // 64*32
  if (e >= 64 * 32) return;
  int co = e >> 5, K = e & 31;
  w016[e] = (f16)(K < 27 ? w0[co * 27 + K] : 0.0f);
}

__global__ void pack_wk(const float* __restrict__ wk, f16* __restrict__ wk16) {
  int e = blockIdx.x * 256 + threadIdx.x;           // 3*64*576, natural flatten
  if (e >= 3 * 64 * 576) return;
  wk16[e] = (f16)wk[e];
}

// S[b][co][t] = sum_c msg[b,c] * wa[co, c, t]   (message-channel fold)
__global__ __launch_bounds__(256)
void msg_fold(const float* __restrict__ msg, const float* __restrict__ wa,
              float* __restrict__ S) {
  __shared__ float red[256];
  int co = blockIdx.x, b = blockIdx.y, tid = threadIdx.x;
  float mv = msg[b * LL + tid];
  for (int t = 0; t < 9; ++t) {
    red[tid] = mv * wa[((size_t)co * 323 + tid) * 9 + t];
    __syncthreads();
    for (int s = 128; s > 0; s >>= 1) {
      if (tid < s) red[tid] += red[tid + s];
      __syncthreads();
    }
    if (tid == 0) S[((size_t)(b * 64 + co)) * 9 + t] = red[0];
    __syncthreads();
  }
}

// Per-batch convA weights: 68 effective channels (enc0, enc1..63, img0..2, ones->S)
__global__ void pack_wA(const float* __restrict__ wa, const float* __restrict__ S,
                        f16* __restrict__ wA16) {
  int e = blockIdx.x * 256 + threadIdx.x;           // 16*64*640
  if (e >= BB * 64 * 640) return;
  int K  = e % 640;
  int bc = e / 640;
  int co = bc & 63;
  int b  = bc >> 6;
  float val = 0.0f;
  if (K < 612) {
    int cp = K / 9, t = K - cp * 9;
    if (cp < 64)      val = wa[((size_t)co * 323 + 256 + cp) * 9 + t];        // enc
    else if (cp < 67) val = wa[((size_t)co * 323 + 320 + (cp - 64)) * 9 + t]; // image
    else              val = S[((size_t)(b * 64 + co)) * 9 + t];               // msg fold
  }
  wA16[e] = (f16)val;
}

// Swizzle [mat][64][Kpad] f16 weights into per-lane A-fragment order:
// out[((mat*4+coTile)*kSteps+s)*512 + lane*16 + e] so each lane reads 32B contiguous.
__global__ void swz_w(const f16* __restrict__ in, f16* __restrict__ out,
                      int Kpad, int nmat) {
  int idx = blockIdx.x * 256 + threadIdx.x;
  int total = nmat * 64 * Kpad;
  if (idx >= total) return;
  int kSteps = Kpad >> 5;
  int e = idx & 15;
  int l = (idx >> 4) & 31;
  int r = idx >> 9;
  int s = r % kSteps;
  int r2 = r / kSteps;
  int t = r2 & 3;
  int mat = r2 >> 2;
  int half = l >> 4, co = t * 16 + (l & 15);
  int v = e >> 1, odd = e & 1;
  int K = s * 32 + (v < 4 ? half * 8 + 2 * v : 16 + half * 8 + 2 * (v - 4)) + odd;
  out[idx] = in[((size_t)mat * 64 + co) * Kpad + K];
}

// ---------------------------------------------------------------------------
// Sparse DFT of plane (b, ch0) at the 256 watermark bins
// ---------------------------------------------------------------------------
__global__ __launch_bounds__(256)
void wm_dft(const float* __restrict__ x, const int* __restrict__ ys,
            const int* __restrict__ xs, float* __restrict__ Fold) {
  __shared__ float twc[256], tws[256];
  __shared__ float rr[256], ri[256];
  int k = blockIdx.x, b = blockIdx.y, tid = threadIdx.x;
  float fs, fc;
  __sincosf(-6.28318530717958647f * (float)tid * (1.0f / 256.0f), &fs, &fc);
  twc[tid] = fc; tws[tid] = fs;
  __syncthreads();
  int yk = ys[k], xk = xs[k];
  const float* xp = x + (size_t)b * CCH * HW;       // channel 0 plane
  float are = 0.f, aim = 0.f;
  for (int it = 0; it < 256; ++it) {
    int p = it * 256 + tid;
    int hh = p >> 8, wp = p & 255;
    int t = (yk * hh + xk * wp) & 255;
    float xv = xp[p];
    are = fmaf(xv, twc[t], are);
    aim = fmaf(xv, tws[t], aim);
  }
  rr[tid] = are; ri[tid] = aim;
  __syncthreads();
  for (int s = 128; s > 0; s >>= 1) {
    if (tid < s) { rr[tid] += rr[tid + s]; ri[tid] += ri[tid + s]; }
    __syncthreads();
  }
  if (tid == 0) {
    Fold[(b * 256 + k) * 2 + 0] = rr[0];
    Fold[(b * 256 + k) * 2 + 1] = ri[0];
  }
}

// enc0 = x0 + (1/N) Re( sum_k ((m+im) - Fold_k) * e^{+i theta} )
__global__ __launch_bounds__(256)
void wm_corr(const float* __restrict__ x, const float* __restrict__ msg,
             const int* __restrict__ ys, const int* __restrict__ xs,
             const float* __restrict__ Fold, float* __restrict__ enc0) {
  __shared__ float twc[256], tws[256];
  __shared__ float dr[256], di[256];
  __shared__ int iy[256], ix[256];
  int chunk = blockIdx.x, b = blockIdx.y, tid = threadIdx.x;
  float fs, fc;
  __sincosf(6.28318530717958647f * (float)tid * (1.0f / 256.0f), &fs, &fc);
  twc[tid] = fc; tws[tid] = fs;
  float mv = msg[b * LL + tid];
  dr[tid] = mv - Fold[(b * 256 + tid) * 2 + 0];
  di[tid] = mv - Fold[(b * 256 + tid) * 2 + 1];
  iy[tid] = ys[tid]; ix[tid] = xs[tid];
  __syncthreads();
  int p = chunk * 256 + tid;
  int hh = p >> 8, wp = p & 255;
  float acc = 0.f;
  for (int k = 0; k < 256; ++k) {
    int t = (iy[k] * hh + ix[k] * wp) & 255;
    acc = fmaf(dr[k], twc[t], acc);
    acc = fmaf(-di[k], tws[t], acc);
  }
  enc0[(size_t)b * HW + p] = x[(size_t)b * CCH * HW + p] + acc * (1.0f / 65536.0f);
}

// ---------------------------------------------------------------------------
// WMMA implicit-GEMM conv3x3 + BN + ReLU.  MODE 0: plain layer (f32 out).
// MODE 1: final layer - gathers (enc0, x1..63, image, ones), per-batch weights,
//         fused 1x1 projection to 3 channels -> d_out.
// Block: 128 threads = 4 waves; wave i computes out-channels [16i,16i+16),
// one row h, a 16-pixel tile. D = A(16co x K) x B(K x 16px).
// Activations staged f32 via async global->LDS, then rearranged once into
// shared B-fragment layout consumed by all 4 waves as ds_load_b128.
// ---------------------------------------------------------------------------
template<int MODE>
__global__ __launch_bounds__(128)
void conv_wmma(const float* __restrict__ xin,
               const float* __restrict__ img,
               const float* __restrict__ enc0,
               const f16*  __restrict__ wgt,      // swizzled A-fragment layout
               const float* __restrict__ bias,
               const float* __restrict__ gam, const float* __restrict__ bet,
               const float* __restrict__ mu,  const float* __restrict__ var,
               const float* __restrict__ wf,  const float* __restrict__ bfin,
               float* __restrict__ out,
               int CinEff, int Kpad)
{
  __shared__ float actf[68 * 54 + 64];     // raw f32 tile [ci][3 rows][18 cols]+zeros
  __shared__ v16h  BldsV[20 * 32];         // B fragments: [kstep][lane] 16 f16 each
  __shared__ unsigned short offK[640];
  __shared__ float sb[64], sg[64], sbe[64], sm[64], sv[64];
  __shared__ float ytile[64 * 16];

  const int tid = threadIdx.x;
  const int w0 = blockIdx.x * 16;
  const int h  = blockIdx.y;
  const int b  = blockIdx.z;
  const int kSteps = Kpad >> 5;

  if (tid < 64) {
    sb[tid] = bias[tid]; sg[tid] = gam[tid]; sbe[tid] = bet[tid];
    sm[tid] = mu[tid];   sv[tid] = var[tid];
  }
  const int Kact  = CinEff * 9;
  const int zbase = CinEff * 54;
  for (int K = tid; K < Kpad; K += 128) {
    unsigned short o;
    if (K < Kact) {
      int ci = K / 9, t = K - ci * 9;
      int kh = t / 3, kw = t - kh * 3;
      o = (unsigned short)(ci * 54 + kh * 18 + kw);
    } else o = (unsigned short)zbase;
    offK[K] = o;
  }
  for (int e = tid; e < 64; e += 128) actf[zbase + e] = 0.0f;

  // --- stage raw f32 activation tile: async copy for in-bounds elements ---
  const int total = CinEff * 54;
  for (int e = tid; e < total; e += 128) {
    int ci = e / 54, rem = e - ci * 54;
    int r = rem / 18, c = rem - r * 18;
    int gy = h - 1 + r, gx = w0 - 1 + c;
    bool inb = (gy >= 0 && gy < HH && gx >= 0 && gx < WWID);
    const float* src = nullptr;
    bool isconst = false; float cval = 0.0f;
    if (inb) {
      size_t sp = (size_t)gy * WWID + gx;
      if (MODE == 0) {
        src = xin + ((size_t)b * CinEff + ci) * HW + sp;
      } else {
        if (ci == 67)      { isconst = true; cval = 1.0f; }                  // ones ch
        else if (ci == 0)  src = enc0 + (size_t)b * HW + sp;                 // watermarked
        else if (ci < 64)  src = xin + ((size_t)b * CCH + ci) * HW + sp;     // enc = x
        else               src = img + ((size_t)b * 3 + (ci - 64)) * HW + sp;
      }
    } else {
      isconst = true; cval = 0.0f;
    }
#if USE_ASYNC
    if (isconst) {
      actf[e] = cval;
    } else {
      unsigned lofs = (unsigned)(size_t)(&actf[e]);   // low 32b = LDS byte offset
      async_g2l_b32(lofs, src);
    }
#else
    actf[e] = isconst ? cval : *src;
#endif
  }
  wait_async0();
  __syncthreads();

  // --- build shared B fragments (f32 -> f16), consumed by all 4 waves ---
  f16* Bf = (f16*)BldsV;
  const int btotal = kSteps * 512;                  // kSteps * 32 lanes * 16 elems
  for (int e = tid; e < btotal; e += 128) {
    int el = e & 15;
    int l  = (e >> 4) & 31;
    int s  = e >> 9;
    int n  = l & 15;
    int K  = s * 32 + ((l >> 4) << 3) + (el < 8 ? el : 8 + el);  // lanes 0-15: K 0..7,16..23; 16-31: 8..15,24..31
    Bf[e] = (f16)actf[offK[K] + n];
  }
  __syncthreads();

  const int wave   = tid >> 5;
  const int lane   = tid & 31;
  const int mrow   = lane & 15;
  const int half   = lane >> 4;
  const int coBase = wave * 16;
  const v16h* wv = (const v16h*)wgt;                // swizzled fragments
  const size_t abase = ((size_t)((MODE == 1 ? b : 0) * 4 + wave) * kSteps) * 32 + lane;

  v8f acc = {};
  #pragma unroll 2
  for (int ks = 0; ks < kSteps; ++ks) {
    v16h afrag = wv[abase + (size_t)ks * 32];       // 2x global_load_b128
    v16h bfrag = BldsV[ks * 32 + lane];             // 2x ds_load_b128
    acc = __builtin_amdgcn_wmma_f32_16x16x32_f16(false, afrag, false, bfrag,
                                                 (short)0, acc, false, false);
  }

  // epilogue: bias + BN + ReLU
  #pragma unroll
  for (int v = 0; v < 8; ++v) {
    int co = coBase + v + 8 * half;                 // D row
    float val = acc[v] + sb[co];
    val = sg[co] * (val - sm[co]) * rsqrtf(sv[co] + EPSV) + sbe[co];
    val = fmaxf(val, 0.0f);
    if (MODE == 0) {
      out[((size_t)b * CCH + co) * HW + (size_t)h * WWID + w0 + mrow] = val;
    } else {
      ytile[co * 16 + mrow] = val;
    }
  }
  if (MODE == 1) {                                  // fused 1x1 projection -> 3 ch
    __syncthreads();
    if (tid < 48) {
      int o = tid >> 4, nn = tid & 15;
      float s = bfin[o];
      #pragma unroll 8
      for (int c = 0; c < 64; ++c) s += wf[o * 64 + c] * ytile[c * 16 + nn];
      out[((size_t)b * 3 + o) * HW + (size_t)h * WWID + w0 + nn] = s;
    }
  }
}

// ---------------------------------------------------------------------------
extern "C" void kernel_launch(void* const* d_in, const int* in_sizes, int n_in,
                              void* d_out, int out_size, void* d_ws, size_t ws_size,
                              hipStream_t stream) {
  const float* image = (const float*)d_in[0];
  const float* msg   = (const float*)d_in[1];
  const float* w0    = (const float*)d_in[2];
  const float* b0    = (const float*)d_in[3];
  const float* g0    = (const float*)d_in[4];
  const float* be0   = (const float*)d_in[5];
  const float* m0    = (const float*)d_in[6];
  const float* v0    = (const float*)d_in[7];
  const float* wk    = (const float*)d_in[8];
  const float* bk    = (const float*)d_in[9];
  const float* gk    = (const float*)d_in[10];
  const float* bek   = (const float*)d_in[11];
  const float* mk    = (const float*)d_in[12];
  const float* vk    = (const float*)d_in[13];
  const float* wa    = (const float*)d_in[14];
  const float* ba    = (const float*)d_in[15];
  const float* ga    = (const float*)d_in[16];
  const float* bea   = (const float*)d_in[17];
  const float* ma    = (const float*)d_in[18];
  const float* va    = (const float*)d_in[19];
  const float* wf    = (const float*)d_in[20];
  const float* bf    = (const float*)d_in[21];
  float* outp = (float*)d_out;

  // workspace carve-up (256B aligned)
  char* ws = (char*)d_ws;
  size_t off = 0;
  auto carve = [&](size_t bytes) { void* p = ws + off; off += (bytes + 255) & ~(size_t)255; return p; };
  float* xA    = (float*)carve((size_t)BB * CCH * HW * 4);
  float* xB    = (float*)carve((size_t)BB * CCH * HW * 4);
  float* enc0  = (float*)carve((size_t)BB * HW * 4);
  f16*   w016  = (f16*)  carve((size_t)64 * 32 * 2);
  f16*   wk16  = (f16*)  carve((size_t)3 * 64 * 576 * 2);
  f16*   wA16  = (f16*)  carve((size_t)BB * 64 * 640 * 2);
  f16*   w0s   = (f16*)  carve((size_t)64 * 32 * 2);
  f16*   wks   = (f16*)  carve((size_t)3 * 64 * 576 * 2);
  f16*   wAs   = (f16*)  carve((size_t)BB * 64 * 640 * 2);
  float* Sbuf  = (float*)carve((size_t)BB * 64 * 9 * 4);
  float* Fold  = (float*)carve((size_t)BB * 256 * 2 * 4);
  int*   ys    = (int*)  carve((size_t)256 * 4);
  int*   xs    = (int*)  carve((size_t)256 * 4);
  (void)ws_size; (void)in_sizes; (void)n_in; (void)out_size;

  // --- prep ---
  wm_idx<<<1, 32, 0, stream>>>(ys, xs);
  pack_w0<<<(64 * 32 + 255) / 256, 256, 0, stream>>>(w0, w016);
  pack_wk<<<(3 * 64 * 576 + 255) / 256, 256, 0, stream>>>(wk, wk16);
  msg_fold<<<dim3(64, BB), 256, 0, stream>>>(msg, wa, Sbuf);
  pack_wA<<<(BB * 64 * 640 + 255) / 256, 256, 0, stream>>>(wa, Sbuf, wA16);
  swz_w<<<(64 * 32 + 255) / 256, 256, 0, stream>>>(w016, w0s, 32, 1);
  swz_w<<<(3 * 64 * 576 + 255) / 256, 256, 0, stream>>>(wk16, wks, 576, 3);
  swz_w<<<(BB * 64 * 640 + 255) / 256, 256, 0, stream>>>(wA16, wAs, 640, BB);

  const dim3 cgrid(WWID / 16, HH, BB);
  const size_t wkMat = (size_t)64 * 576;            // one layer, swizzled size
  // --- conv chain ---
  conv_wmma<0><<<cgrid, 128, 0, stream>>>(image, nullptr, nullptr, w0s,
      b0, g0, be0, m0, v0, nullptr, nullptr, xA, 3, 32);
  conv_wmma<0><<<cgrid, 128, 0, stream>>>(xA, nullptr, nullptr, wks + 0 * wkMat,
      bk + 0, gk + 0, bek + 0, mk + 0, vk + 0, nullptr, nullptr, xB, 64, 576);
  conv_wmma<0><<<cgrid, 128, 0, stream>>>(xB, nullptr, nullptr, wks + 1 * wkMat,
      bk + 64, gk + 64, bek + 64, mk + 64, vk + 64, nullptr, nullptr, xA, 64, 576);
  conv_wmma<0><<<cgrid, 128, 0, stream>>>(xA, nullptr, nullptr, wks + 2 * wkMat,
      bk + 128, gk + 128, bek + 128, mk + 128, vk + 128, nullptr, nullptr, xB, 64, 576);

  // --- watermark: sparse DFT at 256 bins + rank-256 inverse correction ---
  wm_dft<<<dim3(256, BB), 256, 0, stream>>>(xB, ys, xs, Fold);
  wm_corr<<<dim3(256, BB), 256, 0, stream>>>(xB, msg, ys, xs, Fold, enc0);

  // --- final conv (gathered channels, per-batch weights) + fused 1x1 ---
  conv_wmma<1><<<cgrid, 128, 0, stream>>>(xB, image, enc0, wAs,
      ba, ga, bea, ma, va, wf, bf, outp, 68, 640);
}
